// AttentionLayer_7937099563618
// MI455X (gfx1250) — matrix operand order repaired and checked by reading, not compiled, not run on hardware
//
#include <hip/hip_runtime.h>
#include <cstdint>
#include <cstddef>

typedef __attribute__((ext_vector_type(16))) __bf16 v16bf;
typedef __attribute__((ext_vector_type(8)))  float  v8f;

#define B_  4
#define T_  2048
#define C_  1024
#define H_  16
#define D_  64
#define BT_ (B_*T_)

static __device__ __forceinline__ v8f wmma_bf16(v16bf a, v16bf b, v8f c) {
  // D = A(16x32) * B(32x16) + C, f32 accumulate
  return __builtin_amdgcn_wmma_f32_16x16x32_bf16(false, a, false, b, (short)0, c, false, false);
}

// A-matrix fragment (16x32 bf16). Lane l (l<16): M=l, K = {8h..8h+7} U {16+8h..16+8h+7}, h=l>>4.
static __device__ __forceinline__ v16bf load_a_frag(const __bf16* base, int stride, int m0, int k0) {
  const int lane = threadIdx.x & 31;
  const int sub  = lane >> 4;
  const __bf16* p0 = base + (size_t)(m0 + (lane & 15)) * stride + k0 + 8 * sub;
  const __bf16* p1 = p0 + 16;
  v16bf a;
#pragma unroll
  for (int j = 0; j < 8; ++j) { a[j] = p0[j]; a[8 + j] = p1[j]; }
  return a;
}

// B-matrix fragment (32x16 bf16), base indexed [n][k]. Lane l: N=l&15, K = {8h..8h+7} U {16+8h..16+8h+7}, h=l>>4.
static __device__ __forceinline__ v16bf load_b_frag(const __bf16* base, int stride, int n0, int k0) {
  const int lane = threadIdx.x & 31;
  const int sub  = lane >> 4;
  const __bf16* p = base + (size_t)(n0 + (lane & 15)) * stride + k0 + 8 * sub;
  v16bf bfrag;
#pragma unroll
  for (int j = 0; j < 8; ++j) { bfrag[j] = p[j]; bfrag[8 + j] = p[16 + j]; }
  return bfrag;
}

// ---------------------------------------------------------------------------
// Kernel 1: fused QKV projection (x @ W^T) + xPos rotary on Q and K.
// Block = 256 threads (8 waves). Tile: M=128 rows, N=64 cols, K-step 32.
// gridDim.x selects 64-wide column block over stacked [Wq;Wk;Wv] (N=3072).
// ---------------------------------------------------------------------------
__global__ void __launch_bounds__(256) qkv_rope_kernel(
    const float* __restrict__ x, const float* __restrict__ Wq,
    const float* __restrict__ Wk, const float* __restrict__ Wv,
    __bf16* __restrict__ q_ws, __bf16* __restrict__ k_ws, __bf16* __restrict__ v_ws)
{
  __shared__ __bf16 Xs[128][40];   // [m][k], stride 40 bf16 = 80B (16B aligned rows)
  __shared__ __bf16 Wt[64][40];    // [n][k]

  const int tid  = threadIdx.x;
  const int wave = tid >> 5;
  const int lane = tid & 31;
  const int sub  = lane >> 4;
  const int ln   = lane & 15;
  const int nBase = blockIdx.x * 64;        // 0..3071 over stacked QKV
  const int mBase = blockIdx.y * 128;       // 0..8191 row of (B*T)
  const int msel  = nBase >> 10;            // 0=Q, 1=K, 2=V (block-uniform)
  const int nLoc  = nBase & (C_ - 1);
  const float* W  = (msel == 0) ? Wq : (msel == 1) ? Wk : Wv;

  v8f acc[4] = {};

  for (int k0 = 0; k0 < C_; k0 += 32) {
    { // stage X tile 128x32 f32 -> bf16 (each thread: 16 floats)
      const int row = tid & 127;
      const int ko  = (tid >> 7) * 16;
      const float* src = x + (size_t)(mBase + row) * C_ + k0 + ko;
      float4 f0 = ((const float4*)src)[0];
      float4 f1 = ((const float4*)src)[1];
      float4 f2 = ((const float4*)src)[2];
      float4 f3 = ((const float4*)src)[3];
      __bf16* dst = &Xs[row][ko];
      dst[0]=(__bf16)f0.x;  dst[1]=(__bf16)f0.y;  dst[2]=(__bf16)f0.z;  dst[3]=(__bf16)f0.w;
      dst[4]=(__bf16)f1.x;  dst[5]=(__bf16)f1.y;  dst[6]=(__bf16)f1.z;  dst[7]=(__bf16)f1.w;
      dst[8]=(__bf16)f2.x;  dst[9]=(__bf16)f2.y;  dst[10]=(__bf16)f2.z; dst[11]=(__bf16)f2.w;
      dst[12]=(__bf16)f3.x; dst[13]=(__bf16)f3.y; dst[14]=(__bf16)f3.z; dst[15]=(__bf16)f3.w;
    }
    { // stage W tile 64x32 f32 -> bf16 (each thread: 8 floats)
      const int row = tid & 63;
      const int ko  = (tid >> 6) * 8;
      const float* src = W + (size_t)(nLoc + row) * C_ + k0 + ko;
      float4 f0 = ((const float4*)src)[0];
      float4 f1 = ((const float4*)src)[1];
      __bf16* dst = &Wt[row][ko];
      dst[0]=(__bf16)f0.x; dst[1]=(__bf16)f0.y; dst[2]=(__bf16)f0.z; dst[3]=(__bf16)f0.w;
      dst[4]=(__bf16)f1.x; dst[5]=(__bf16)f1.y; dst[6]=(__bf16)f1.z; dst[7]=(__bf16)f1.w;
    }
    __syncthreads();
    const v16bf a = load_a_frag(&Xs[0][0], 40, wave * 16, 0);
#pragma unroll
    for (int j = 0; j < 4; ++j) {
      const v16bf bf = load_b_frag(&Wt[0][0], 40, j * 16, 0);
      acc[j] = wmma_bf16(a, bf, acc[j]);
    }
    __syncthreads();
  }

  // Epilogue: C layout => row = mBase + wave*16 + 8*sub + r, col = nLoc + j*16 + ln.
  if (msel == 2) {
#pragma unroll
    for (int j = 0; j < 4; ++j) {
      const int n = nLoc + j * 16 + ln;
      const int h = n >> 6, d = n & 63;
#pragma unroll
      for (int r = 0; r < 8; ++r) {
        const int gr = mBase + wave * 16 + 8 * sub + r;
        const int bb = gr >> 11, t = gr & (T_ - 1);
        v_ws[(((size_t)bb * H_ + h) * T_ + t) * D_ + d] = (__bf16)acc[j][r];
      }
    }
  } else {
    __bf16* outp = (msel == 0) ? q_ws : k_ws;
    const float esign = (msel == 0) ? 1.0f : -1.0f;   // q *= scale ; k /= scale
#pragma unroll
    for (int j = 0; j < 4; ++j) {
      const int n = nLoc + j * 16 + ln;
      const int h = n >> 6, d = n & 63;
      const int dpair = d & ~1;
      // inv_freq = 10000^(-dpair/64); log2(10000)=13.2877123795
      const float invf = __builtin_amdgcn_exp2f(-(float)dpair * (13.287712379549449f / 64.0f));
      const float l2b  = __log2f(((float)dpair + 25.6f) * (1.0f / 89.6f));
      const bool even  = (d & 1) == 0;
#pragma unroll
      for (int r = 0; r < 8; ++r) {
        const int gr = mBase + wave * 16 + 8 * sub + r;
        const int bb = gr >> 11, t = gr & (T_ - 1);
        const float v = acc[j][r];
        const float p = __shfl_xor(v, 1);             // pair element d^1 (adjacent lane)
        const float ang = (float)t * invf;
        const float cs = __cosf(ang), sn = __sinf(ang);
        const float rot = even ? (v * cs - p * sn) : (v * cs + p * sn);
        const float pw  = ((float)t - (float)(T_ / 2)) * (1.0f / 512.0f);
        const float sc  = __builtin_amdgcn_exp2f(esign * l2b * pw);
        outp[(((size_t)bb * H_ + h) * T_ + t) * D_ + d] = (__bf16)(rot * sc);
      }
    }
  }
}

// ---------------------------------------------------------------------------
// Kernel 2: flash attention. Block = 128 threads (4 waves), 64 q-rows/block,
// online softmax over 32-wide K blocks. One (b,h) per blockIdx.y.
// ---------------------------------------------------------------------------
__global__ void __launch_bounds__(128) attention_kernel(
    const __bf16* __restrict__ qw, const __bf16* __restrict__ kw,
    const __bf16* __restrict__ vw, const int* __restrict__ mask,
    __bf16* __restrict__ attn)
{
  __shared__ __bf16 Qs[64][72];      // [t][d]
  __shared__ __bf16 Ks[32][72];      // [k-row][d]  -> natural B-layout for S = Q K^T
  __shared__ __bf16 VsT[64][48];     // [d][k-row]  -> B-layout for O = P V
  __shared__ __bf16 Ps[4][16][48];   // per-wave P tile (16 x 32)

  const int tid  = threadIdx.x;
  const int wave = tid >> 5;
  const int lane = tid & 31;
  const int sub  = lane >> 4;
  const int ln   = lane & 15;
  const int bh   = blockIdx.y;
  const int b    = bh >> 4;
  const int h    = bh & 15;
  const int qbase = blockIdx.x * 64;

  const __bf16* qptr = qw + ((size_t)bh * T_ + qbase) * D_;
  const __bf16* kmat = kw + (size_t)bh * T_ * D_;
  const __bf16* vmat = vw + (size_t)bh * T_ * D_;

  for (int i = tid; i < 64 * 8; i += 128) {
    const int row = i >> 3, seg = i & 7;
    *(uint4*)&Qs[row][seg * 8] = *(const uint4*)(qptr + (size_t)row * D_ + seg * 8);
  }
  __syncthreads();

  const v16bf aq0 = load_a_frag(&Qs[0][0], 72, wave * 16, 0);
  const v16bf aq1 = load_a_frag(&Qs[0][0], 72, wave * 16, 32);

  v8f O[4] = {};
  float mrow[8], lrow[8];
#pragma unroll
  for (int r = 0; r < 8; ++r) { mrow[r] = -1e30f; lrow[r] = 0.0f; }

  for (int kb = 0; kb < T_; kb += 32) {
    __syncthreads();   // protect previous iteration's Ks/VsT fragment reads
    for (int i = tid; i < 32 * 8; i += 128) {
      const int row = i >> 3, seg = i & 7;
      *(uint4*)&Ks[row][seg * 8] = *(const uint4*)(kmat + (size_t)(kb + row) * D_ + seg * 8);
    }
    for (int i = tid; i < 32 * 8; i += 128) {
      const int row = i >> 3, seg = i & 7;
      __bf16 tmp[8];
      *(uint4*)tmp = *(const uint4*)(vmat + (size_t)(kb + row) * D_ + seg * 8);
#pragma unroll
      for (int jj = 0; jj < 8; ++jj) VsT[seg * 8 + jj][row] = tmp[jj];
    }
    __syncthreads();

    // S tiles: 16(q) x 32(k), contraction over D=64 in two K=32 WMMA steps.
    v8f s0 = {}, s1 = {};
    {
      v16bf bk = load_b_frag(&Ks[0][0], 72, 0, 0);
      s0 = wmma_bf16(aq0, bk, s0);
      bk = load_b_frag(&Ks[0][0], 72, 0, 32);
      s0 = wmma_bf16(aq1, bk, s0);
      bk = load_b_frag(&Ks[0][0], 72, 16, 0);
      s1 = wmma_bf16(aq0, bk, s1);
      bk = load_b_frag(&Ks[0][0], 72, 16, 32);
      s1 = wmma_bf16(aq1, bk, s1);
    }

    const int tq0 = qbase + wave * 16 + 8 * sub;
#pragma unroll
    for (int r = 0; r < 8; ++r) {
      const int tq = tq0 + r;
      const int* mr = mask + ((size_t)b * T_ + tq) * T_ + kb;
      float v0 = s0[r] * 0.03125f;                 // 1/sqrt(C) = 1/32
      float v1 = s1[r] * 0.03125f;
      if (mr[ln] == 0)      v0 = -1e4f;
      if (mr[16 + ln] == 0) v1 = -1e4f;
      float mx = fmaxf(v0, v1);
#pragma unroll
      for (int xm = 1; xm < 16; xm <<= 1) mx = fmaxf(mx, __shfl_xor(mx, xm));
      const float mnew = fmaxf(mrow[r], mx);
      const float p0 = __expf(v0 - mnew);
      const float p1 = __expf(v1 - mnew);
      float rs = p0 + p1;
#pragma unroll
      for (int xm = 1; xm < 16; xm <<= 1) rs += __shfl_xor(rs, xm);
      const float alpha = __expf(mrow[r] - mnew);
      lrow[r] = lrow[r] * alpha + rs;
      mrow[r] = mnew;
#pragma unroll
      for (int j = 0; j < 4; ++j) O[j][r] *= alpha;
      Ps[wave][8 * sub + r][ln]      = (__bf16)p0;
      Ps[wave][8 * sub + r][16 + ln] = (__bf16)p1;
    }

    // Wave-private P buffer: only intra-wave DS ordering needed.
    asm volatile("s_wait_dscnt 0" ::: "memory");

    const v16bf ap = load_a_frag(&Ps[wave][0][0], 48, 0, 0);
#pragma unroll
    for (int j = 0; j < 4; ++j) {
      const v16bf bv = load_b_frag(&VsT[0][0], 48, j * 16, 0);
      O[j] = wmma_bf16(ap, bv, O[j]);
    }
  }

#pragma unroll
  for (int j = 0; j < 4; ++j) {
#pragma unroll
    for (int r = 0; r < 8; ++r) {
      const int t = qbase + wave * 16 + 8 * sub + r;
      const int c = h * D_ + j * 16 + ln;
      attn[((size_t)b * T_ + t) * C_ + c] = (__bf16)(O[j][r] / lrow[r]);
    }
  }
}

// ---------------------------------------------------------------------------
// Kernel 3: output projection attn(bf16) @ Wo^T -> f32 out.
// ---------------------------------------------------------------------------
__global__ void __launch_bounds__(256) out_proj_kernel(
    const __bf16* __restrict__ attn, const float* __restrict__ Wo,
    float* __restrict__ out)
{
  __shared__ __bf16 Xs[128][40];
  __shared__ __bf16 Wt[64][40];

  const int tid  = threadIdx.x;
  const int wave = tid >> 5;
  const int lane = tid & 31;
  const int sub  = lane >> 4;
  const int ln   = lane & 15;
  const int nBase = blockIdx.x * 64;
  const int mBase = blockIdx.y * 128;

  v8f acc[4] = {};

  for (int k0 = 0; k0 < C_; k0 += 32) {
    for (int i = tid; i < 512; i += 256) {   // 128x32 bf16, 16B chunks
      const int row = i >> 2, seg = i & 3;
      *(uint4*)&Xs[row][seg * 8] =
          *(const uint4*)(attn + (size_t)(mBase + row) * C_ + k0 + seg * 8);
    }
    {
      const int row = tid & 63;
      const int ko  = (tid >> 6) * 8;
      const float* src = Wo + (size_t)(nBase + row) * C_ + k0 + ko;
      float4 f0 = ((const float4*)src)[0];
      float4 f1 = ((const float4*)src)[1];
      __bf16* dst = &Wt[row][ko];
      dst[0]=(__bf16)f0.x; dst[1]=(__bf16)f0.y; dst[2]=(__bf16)f0.z; dst[3]=(__bf16)f0.w;
      dst[4]=(__bf16)f1.x; dst[5]=(__bf16)f1.y; dst[6]=(__bf16)f1.z; dst[7]=(__bf16)f1.w;
    }
    __syncthreads();
    const v16bf a = load_a_frag(&Xs[0][0], 40, wave * 16, 0);
#pragma unroll
    for (int j = 0; j < 4; ++j) {
      const v16bf bf = load_b_frag(&Wt[0][0], 40, j * 16, 0);
      acc[j] = wmma_bf16(a, bf, acc[j]);
    }
    __syncthreads();
  }

#pragma unroll
  for (int j = 0; j < 4; ++j) {
    const int n = nBase + j * 16 + ln;
#pragma unroll
    for (int r = 0; r < 8; ++r) {
      const int gr = mBase + wave * 16 + 8 * sub + r;
      out[(size_t)gr * C_ + n] = acc[j][r];
    }
  }
}

// ---------------------------------------------------------------------------
extern "C" void kernel_launch(void* const* d_in, const int* in_sizes, int n_in,
                              void* d_out, int out_size, void* d_ws, size_t ws_size,
                              hipStream_t stream) {
  (void)in_sizes; (void)n_in; (void)out_size; (void)ws_size;
  const float* x    = (const float*)d_in[0];
  const int*   mask = (const int*)d_in[1];
  const float* Wq   = (const float*)d_in[2];
  const float* Wk   = (const float*)d_in[3];
  const float* Wv   = (const float*)d_in[4];
  const float* Wo   = (const float*)d_in[5];
  float* out = (float*)d_out;

  const size_t per = (size_t)B_ * H_ * T_ * D_;   // 8,388,608 elements
  __bf16* q_ws    = (__bf16*)d_ws;
  __bf16* k_ws    = q_ws + per;
  __bf16* v_ws    = k_ws + per;
  __bf16* attn_ws = v_ws + per;                   // total 64 MB bf16 scratch

  qkv_rope_kernel<<<dim3(3 * C_ / 64, BT_ / 128), 256, 0, stream>>>(
      x, Wq, Wk, Wv, q_ws, k_ws, v_ws);
  attention_kernel<<<dim3(T_ / 64, B_ * H_), 128, 0, stream>>>(
      q_ws, k_ws, v_ws, mask, attn_ws);
  out_proj_kernel<<<dim3(C_ / 64, BT_ / 128), 256, 0, stream>>>(
      attn_ws, Wo, out);
}
